// Sequence_17927193493537
// MI455X (gfx1250) — compile-verified, matrix operand order checked by reading, not run on hardware
//
#include <hip/hip_runtime.h>
#include <hip/hip_bf16.h>

// ---------------------------------------------------------------------------
// Persistent 2-layer GRU on gfx1250 using v_wmma_f32_16x16x32_bf16.
//   - 32 workgroups, each owns 32 batch rows for all 1024 timesteps.
//   - Hidden state (h1,h2) double-buffered in LDS as bf16 (padded stride).
//   - Weights pre-swizzled once into WMMA B-fragment layout (bf16) in d_ws.
//   - Each wave owns matching r/z/n gate columns -> gate math fused in VGPRs.
// ---------------------------------------------------------------------------

typedef __bf16 bf16_t;
typedef bf16_t v16bf __attribute__((ext_vector_type(16)));
typedef bf16_t v8bf  __attribute__((ext_vector_type(8)));
typedef float  v8f   __attribute__((ext_vector_type(8)));

#define HDIM     512
#define T_TOTAL  1024
#define T_WARM   512
#define MTILE    32            // batch rows per workgroup
#define SHSTR    520           // padded LDS row stride (bf16 elems) -> bank spread
#define NTHREADS 256
#define JBW      4             // j-blocks (of 16 cols) per wave: 32 jb / 8 waves
#define BUFE     (MTILE*SHSTR) // elems per state buffer

__device__ __forceinline__ v8f wmma_bf16(v16bf a, v16bf b, v8f c) {
    return __builtin_amdgcn_wmma_f32_16x16x32_bf16(
        /*neg_a=*/false, a, /*neg_b=*/false, b,
        /*c_mod=*/(short)0, c, /*reuse_a=*/false, /*reuse_b=*/false);
}

// A-fragment (16x32 bf16) for one 16-row tile from LDS state.
// Lane (hi,lo): M=lo. elem i -> K = kt*32 + 8*hi + i        (i<8)
//                          K = kt*32 + 16 + 8*hi + (i-8)    (i>=8)
// => two contiguous 16-byte ds loads per lane.
__device__ __forceinline__ v16bf load_a_frag(const bf16_t* row, int kt, int hi) {
    const bf16_t* p = row + kt * 32 + hi * 8;
    union { v16bf v; v8bf h[2]; } u;
    u.h[0] = *(const v8bf*)(p);
    u.h[1] = *(const v8bf*)(p + 16);
    return u.v;
}

// Packed B-fragment base: layout [slot][jb][gate][kt][lane][16] bf16
__device__ __forceinline__ const bf16_t* wfrag(const bf16_t* wp, int slot, int jb,
                                               int g, int lane) {
    return wp + (size_t)(((slot * 32 + jb) * 3 + g) * 16) * 512 + lane * 16;
}

__device__ __forceinline__ float sigm(float x) { return 1.0f / (1.0f + __expf(-x)); }
__device__ __forceinline__ float tanh_fast(float x) {
    return 2.0f / (1.0f + __expf(-2.0f * x)) - 1.0f;
}

// ---------------------------------------------------------------------------
// Prep: swizzle one [1536,512] f32 weight matrix into bf16 WMMA B-fragments.
// B-fragment (32x16): lane (hi,lo) holds column n=lo, K mapping mirrors A.
// ---------------------------------------------------------------------------
__global__ void pack_weights_kernel(const float* __restrict__ W,
                                    bf16_t* __restrict__ dst) {
    int gidx = blockIdx.x * blockDim.x + threadIdx.x;   // one thread per (frag,lane)
    if (gidx >= 1536 * 32) return;
    const int lane = gidx & 31;
    const int frag = gidx >> 5;          // (jb*3+g)*16 + kt
    const int kt   = frag & 15;
    const int nt   = frag >> 4;          // jb*3+g
    const int g    = nt % 3;
    const int jb   = nt / 3;
    const int lo   = lane & 15;
    const int hi   = lane >> 4;
    const int row  = g * HDIM + jb * 16 + lo;           // output-gate row of W
    v16bf val;
#pragma unroll
    for (int i = 0; i < 16; ++i) {
        const int k = kt * 32 + 8 * hi + ((i < 8) ? i : (i + 8));
        val[i] = (bf16_t)W[(size_t)row * HDIM + k];
    }
    *(v16bf*)(dst + (size_t)frag * 512 + lane * 16) = val;
}

// ---------------------------------------------------------------------------
// Persistent GRU sequence kernel.
// ---------------------------------------------------------------------------
__global__ __launch_bounds__(NTHREADS, 1)
void gru_seq_kernel(const float* __restrict__ input,   // [1024,1024]
                    const float* __restrict__ wih1,    // [1536]
                    const float* __restrict__ bih1,    // [1536]
                    const float* __restrict__ bhh1,    // [1536]
                    const float* __restrict__ bih2,    // [1536]
                    const float* __restrict__ bhh2,    // [1536]
                    const float* __restrict__ wlin,    // [512]
                    const float* __restrict__ blin,    // [1]
                    const bf16_t* __restrict__ wp,     // packed weights, 3 slots
                    float* __restrict__ out)           // [1024,1024]
{
    extern __shared__ char smem_raw[];
    bf16_t* sh1  = (bf16_t*)smem_raw;        // [2][MTILE][SHSTR]
    bf16_t* sh2  = sh1 + 2 * BUFE;           // [2][MTILE][SHSTR]
    float*  sx   = (float*)(sh2 + 2 * BUFE); // [MTILE] current input x
    float*  sout = sx + MTILE;               // [MTILE] readout accumulator

    const int tid  = threadIdx.x;
    const int lane = tid & 31;
    const int wv   = tid >> 5;
    const int lo   = lane & 15;
    const int hi   = lane >> 4;
    const int bm   = blockIdx.x * MTILE;     // first batch row of this WG

    for (int i = tid; i < 2 * BUFE; i += NTHREADS) {
        sh1[i] = (bf16_t)0.0f;
        sh2[i] = (bf16_t)0.0f;
    }
    if (tid < MTILE) sout[tid] = 0.0f;
    __syncthreads();

    const float blin0 = blin[0];
    int cur = 0;

#pragma unroll 1
    for (int t = 0; t < T_TOTAL; ++t) {
        const int nxt = cur ^ 1;
        if (tid < MTILE) {
            const float xv = (t < T_WARM)
                                 ? input[(size_t)(bm + tid) * T_TOTAL + t]
                                 : sout[tid];        // closed-loop feedback
            sx[tid]   = xv;
            sout[tid] = 0.0f;
        }
        __syncthreads();

        bf16_t* h1c = sh1 + cur * BUFE;
        bf16_t* h1n = sh1 + nxt * BUFE;
        bf16_t* h2c = sh2 + cur * BUFE;
        bf16_t* h2n = sh2 + nxt * BUFE;

        // ------------------- Layer 1: gh = h1 @ W_hh1^T -------------------
#pragma unroll 1
        for (int q = 0; q < JBW; ++q) {
            const int jb = wv * JBW + q;
            v8f acc[3][2] = {};
            const bf16_t* w0  = wfrag(wp, 0, jb, 0, lane);
            const bf16_t* w1  = wfrag(wp, 0, jb, 1, lane);
            const bf16_t* w2  = wfrag(wp, 0, jb, 2, lane);
            const bf16_t* ar0 = h1c + lo * SHSTR;
            const bf16_t* ar1 = h1c + (16 + lo) * SHSTR;
#pragma unroll 4
            for (int kt = 0; kt < 16; ++kt) {
                v16bf a0 = load_a_frag(ar0, kt, hi);
                v16bf a1 = load_a_frag(ar1, kt, hi);
                v16bf b0 = *(const v16bf*)(w0 + (size_t)kt * 512);
                v16bf b1 = *(const v16bf*)(w1 + (size_t)kt * 512);
                v16bf b2 = *(const v16bf*)(w2 + (size_t)kt * 512);
                acc[0][0] = wmma_bf16(a0, b0, acc[0][0]);
                acc[0][1] = wmma_bf16(a1, b0, acc[0][1]);
                acc[1][0] = wmma_bf16(a0, b1, acc[1][0]);
                acc[1][1] = wmma_bf16(a1, b1, acc[1][1]);
                acc[2][0] = wmma_bf16(a0, b2, acc[2][0]);
                acc[2][1] = wmma_bf16(a1, b2, acc[2][1]);
            }
            const int j   = jb * 16 + lo;
            const float wR = wih1[j], wZ = wih1[HDIM + j], wN = wih1[2 * HDIM + j];
            const float iR = bih1[j], iZ = bih1[HDIM + j], iN = bih1[2 * HDIM + j];
            const float hR = bhh1[j], hZ = bhh1[HDIM + j], hN = bhh1[2 * HDIM + j];
#pragma unroll
            for (int mt = 0; mt < 2; ++mt) {
#pragma unroll
                for (int r2 = 0; r2 < 8; ++r2) {
                    const int   m  = mt * 16 + r2 + 8 * hi;  // D-frag row map
                    const float xv = sx[m];
                    const float rg = sigm(xv * wR + iR + acc[0][mt][r2] + hR);
                    const float zg = sigm(xv * wZ + iZ + acc[1][mt][r2] + hZ);
                    const float ng = tanh_fast(xv * wN + iN + rg * (acc[2][mt][r2] + hN));
                    const float hp = (float)h1c[m * SHSTR + j];
                    const float hv = (1.0f - zg) * ng + zg * hp;
                    h1n[m * SHSTR + j] = (bf16_t)hv;
                }
            }
        }
        __syncthreads();

        // --------- Layer 2: gi = h1' @ W_ih2^T, gh = h2 @ W_hh2^T ---------
        float oacc[2][8] = {};
#pragma unroll 1
        for (int q = 0; q < JBW; ++q) {
            const int jb = wv * JBW + q;
            v8f gi[3][2] = {};
            v8f gh[3][2] = {};
            {   // gi pass: A = new h1
                const bf16_t* w0  = wfrag(wp, 1, jb, 0, lane);
                const bf16_t* w1  = wfrag(wp, 1, jb, 1, lane);
                const bf16_t* w2  = wfrag(wp, 1, jb, 2, lane);
                const bf16_t* ar0 = h1n + lo * SHSTR;
                const bf16_t* ar1 = h1n + (16 + lo) * SHSTR;
#pragma unroll 4
                for (int kt = 0; kt < 16; ++kt) {
                    v16bf a0 = load_a_frag(ar0, kt, hi);
                    v16bf a1 = load_a_frag(ar1, kt, hi);
                    v16bf b0 = *(const v16bf*)(w0 + (size_t)kt * 512);
                    v16bf b1 = *(const v16bf*)(w1 + (size_t)kt * 512);
                    v16bf b2 = *(const v16bf*)(w2 + (size_t)kt * 512);
                    gi[0][0] = wmma_bf16(a0, b0, gi[0][0]);
                    gi[0][1] = wmma_bf16(a1, b0, gi[0][1]);
                    gi[1][0] = wmma_bf16(a0, b1, gi[1][0]);
                    gi[1][1] = wmma_bf16(a1, b1, gi[1][1]);
                    gi[2][0] = wmma_bf16(a0, b2, gi[2][0]);
                    gi[2][1] = wmma_bf16(a1, b2, gi[2][1]);
                }
            }
            {   // gh pass: A = current h2
                const bf16_t* w0  = wfrag(wp, 2, jb, 0, lane);
                const bf16_t* w1  = wfrag(wp, 2, jb, 1, lane);
                const bf16_t* w2  = wfrag(wp, 2, jb, 2, lane);
                const bf16_t* ar0 = h2c + lo * SHSTR;
                const bf16_t* ar1 = h2c + (16 + lo) * SHSTR;
#pragma unroll 4
                for (int kt = 0; kt < 16; ++kt) {
                    v16bf a0 = load_a_frag(ar0, kt, hi);
                    v16bf a1 = load_a_frag(ar1, kt, hi);
                    v16bf b0 = *(const v16bf*)(w0 + (size_t)kt * 512);
                    v16bf b1 = *(const v16bf*)(w1 + (size_t)kt * 512);
                    v16bf b2 = *(const v16bf*)(w2 + (size_t)kt * 512);
                    gh[0][0] = wmma_bf16(a0, b0, gh[0][0]);
                    gh[0][1] = wmma_bf16(a1, b0, gh[0][1]);
                    gh[1][0] = wmma_bf16(a0, b1, gh[1][0]);
                    gh[1][1] = wmma_bf16(a1, b1, gh[1][1]);
                    gh[2][0] = wmma_bf16(a0, b2, gh[2][0]);
                    gh[2][1] = wmma_bf16(a1, b2, gh[2][1]);
                }
            }
            const int j   = jb * 16 + lo;
            const float iR = bih2[j], iZ = bih2[HDIM + j], iN = bih2[2 * HDIM + j];
            const float hR = bhh2[j], hZ = bhh2[HDIM + j], hN = bhh2[2 * HDIM + j];
            const float wl = wlin[j];
#pragma unroll
            for (int mt = 0; mt < 2; ++mt) {
#pragma unroll
                for (int r2 = 0; r2 < 8; ++r2) {
                    const int   m  = mt * 16 + r2 + 8 * hi;
                    const float rg = sigm(gi[0][mt][r2] + iR + gh[0][mt][r2] + hR);
                    const float zg = sigm(gi[1][mt][r2] + iZ + gh[1][mt][r2] + hZ);
                    const float ng = tanh_fast(gi[2][mt][r2] + iN + rg * (gh[2][mt][r2] + hN));
                    const float hp = (float)h2c[m * SHSTR + j];
                    const float hv = (1.0f - zg) * ng + zg * hp;
                    h2n[m * SHSTR + j] = (bf16_t)hv;
                    oacc[mt][r2] += hv * wl;      // readout partial
                }
            }
        }

        // ----- readout: butterfly over 16 N-lanes, then LDS atomic -----
#pragma unroll
        for (int mt = 0; mt < 2; ++mt) {
#pragma unroll
            for (int r2 = 0; r2 < 8; ++r2) {
                float v = oacc[mt][r2];
                v += __shfl_xor(v, 1, 32);
                v += __shfl_xor(v, 2, 32);
                v += __shfl_xor(v, 4, 32);
                v += __shfl_xor(v, 8, 32);
                if (lo == 0) atomicAdd(&sout[mt * 16 + r2 + 8 * hi], v);
            }
        }
        __syncthreads();

        if (tid < MTILE) {
            const float o = sout[tid] + blin0;
            out[(size_t)(bm + tid) * T_TOTAL + t] = o;
            sout[tid] = o;   // next-step autoregressive input
        }
        cur = nxt;
    }
}

// ---------------------------------------------------------------------------
extern "C" void kernel_launch(void* const* d_in, const int* in_sizes, int n_in,
                              void* d_out, int out_size, void* d_ws, size_t ws_size,
                              hipStream_t stream) {
    const float* input = (const float*)d_in[0];
    const float* W_ih1 = (const float*)d_in[1];
    const float* W_hh1 = (const float*)d_in[2];
    const float* b_ih1 = (const float*)d_in[3];
    const float* b_hh1 = (const float*)d_in[4];
    const float* W_ih2 = (const float*)d_in[5];
    const float* W_hh2 = (const float*)d_in[6];
    const float* b_ih2 = (const float*)d_in[7];
    const float* b_hh2 = (const float*)d_in[8];
    const float* W_lin = (const float*)d_in[9];
    const float* b_lin = (const float*)d_in[10];

    bf16_t* wp = (bf16_t*)d_ws;
    const size_t slotElems = (size_t)1536 * 512;

    // Swizzle the three large weight matrices into WMMA B-fragment layout.
    const int packThreads = 1536 * 32;
    const int packBlocks  = (packThreads + 255) / 256;
    pack_weights_kernel<<<packBlocks, 256, 0, stream>>>(W_hh1, wp + 0 * slotElems);
    pack_weights_kernel<<<packBlocks, 256, 0, stream>>>(W_ih2, wp + 1 * slotElems);
    pack_weights_kernel<<<packBlocks, 256, 0, stream>>>(W_hh2, wp + 2 * slotElems);

    const size_t smemBytes = (size_t)4 * BUFE * sizeof(bf16_t) + 2 * MTILE * sizeof(float);
    (void)hipFuncSetAttribute((const void*)gru_seq_kernel,
                              hipFuncAttributeMaxDynamicSharedMemorySize,
                              (int)smemBytes);

    gru_seq_kernel<<<1024 / MTILE, NTHREADS, smemBytes, stream>>>(
        input, W_ih1, b_ih1, b_hh1, b_ih2, b_hh2, W_lin, b_lin, wp, (float*)d_out);
}
